// VisionModel_6476810683043
// MI455X (gfx1250) — compile-verified, matrix operand order checked by reading, not run on hardware
//
#include <hip/hip_runtime.h>
#include <math.h>

// ---------------- model constants ----------------
constexpr int NT   = 2048;   // tokens
constexpr int DM   = 1280;   // hidden
constexpr int NH   = 16;     // heads
constexpr int KD   = 80;     // head dim
constexpr int KSTR = 128;    // padded head-dim stride (zero-filled 80..127)
constexpr int NL   = 4;      // layers
constexpr int FFD  = 5120;   // mlp dim
constexpr int GG   = 48;     // pos grid
constexpr int OUTD = 2048;   // merger out
constexpr int IMG  = 512;    // tokens per image (4 images of [1,16,32])
constexpr int NIMG = 4;

// ---------------- vector types ----------------
typedef __attribute__((ext_vector_type(16))) __bf16 bf16x16;
typedef __attribute__((ext_vector_type(8)))  __bf16 bf16x8;
typedef __attribute__((ext_vector_type(4)))  __bf16 bf16x4;
typedef __attribute__((ext_vector_type(8)))  float  f32x8;

__device__ inline float gelu_t(float x) {
  float x3 = x * x * x;
  return 0.5f * x * (1.0f + tanhf(0.7978845608028654f * (x + 0.044715f * x3)));
}

// =====================================================================
// Generic batched GEMM:  C = epilogue( (A @ B) * scale + bias ) [+resid]
// A: fp32 row-major (m,k) = A[m*lda + k]       -- M, K assumed tile-aligned
// B: fp32 strided   (k,n) = B[k*bsk + n*bsn]   -- padded so n0..n0+127 readable
// fp32 -> bf16 in LDS (double buffered), v_wmma_f32_16x16x32_bf16 compute.
// Block tile 128x128x32, 256 threads = 8 waves, wave tile 32x64 (2x4 frags).
// Loads are UNGUARDED: caller guarantees padded/aligned extents.
// Epilogue guards only n < Nd (PV gemm stores 80 of 128 columns).
// =====================================================================
#define BM 128
#define BN 128
#define BK 32
#define LSTR 40   // bf16 elems per LDS row (32 data + 8 pad) -> 80B, 16B aligned

__device__ __forceinline__ void fill_tiles(
    __bf16* __restrict__ sA, __bf16* __restrict__ sB,
    const float* __restrict__ A, long lda,
    const float* __restrict__ B, long bsk, long bsn,
    int m0, int n0, int k0, int tid, bool brow)
{
#pragma unroll
  for (int it = 0; it < 4; ++it) {          // A: 128x32 fp32, 4 float4 per thread
    int qq = tid + it * 256;
    int r  = qq >> 3;
    int c4 = (qq & 7) * 4;
    float4 f = *(const float4*)(A + (long)(m0 + r) * lda + k0 + c4);
    bf16x4 v4;
    v4[0] = (__bf16)f.x; v4[1] = (__bf16)f.y; v4[2] = (__bf16)f.z; v4[3] = (__bf16)f.w;
    *(bf16x4*)(&sA[r * LSTR + c4]) = v4;
  }
  if (brow) {                               // B row-major: coalesced over n, scatter to sB[n][k]
#pragma unroll
    for (int it = 0; it < 4; ++it) {
      int qq = tid + it * 256;
      int kk = qq >> 5;            // 0..31
      int n4 = (qq & 31) * 4;      // 0..124
      float4 f = *(const float4*)(B + (long)(k0 + kk) * bsk + n0 + n4);
      sB[(n4 + 0) * LSTR + kk] = (__bf16)f.x;
      sB[(n4 + 1) * LSTR + kk] = (__bf16)f.y;
      sB[(n4 + 2) * LSTR + kk] = (__bf16)f.z;
      sB[(n4 + 3) * LSTR + kk] = (__bf16)f.w;
    }
  } else {                                  // B column-major (K^T): coalesced over k
#pragma unroll
    for (int it = 0; it < 4; ++it) {
      int qq = tid + it * 256;
      int r  = qq >> 3;
      int k4 = (qq & 7) * 4;
      float4 f = *(const float4*)(B + (long)(n0 + r) * bsn + k0 + k4);
      bf16x4 v4;
      v4[0] = (__bf16)f.x; v4[1] = (__bf16)f.y; v4[2] = (__bf16)f.z; v4[3] = (__bf16)f.w;
      *(bf16x4*)(&sB[r * LSTR + k4]) = v4;
    }
  }
}

__launch_bounds__(256)
__global__ void gemm_wmma_bf16(
    const float* __restrict__ A, long lda,
    const float* __restrict__ B, long bsk, long bsn,
    const float* __restrict__ bias,
    const float* __restrict__ resid, long ldr,
    float* __restrict__ C, long ldc,
    int Nd, int Kd,
    float scale, int act,
    int nb2,
    long ba1, long ba2, long bb1, long bb2,
    long bc1, long bc2, long br1, long br2)
{
  __shared__ __align__(16) __bf16 sA[2][BM * LSTR];
  __shared__ __align__(16) __bf16 sB[2][BN * LSTR];

  const int bz = blockIdx.z;
  const int b1 = bz / nb2, b2 = bz % nb2;
  A += (long)b1 * ba1 + (long)b2 * ba2;
  B += (long)b1 * bb1 + (long)b2 * bb2;
  C += (long)b1 * bc1 + (long)b2 * bc2;
  if (resid) resid += (long)b1 * br1 + (long)b2 * br2;

  const int m0   = blockIdx.y * BM;
  const int n0   = blockIdx.x * BN;
  const int tid  = threadIdx.x;
  const int lane = tid & 31;
  const int wid  = tid >> 5;
  const int wmi  = wid >> 1;   // 0..3 : 32-row slab
  const int wni  = wid & 1;    // 0..1 : 64-col slab
  const int l16  = lane & 15;
  const int hsel = lane >> 4;  // 0/1 -> K-pattern half
  const bool brow = (bsn == 1);

  f32x8 acc[2][4];
#pragma unroll
  for (int i = 0; i < 2; ++i)
#pragma unroll
    for (int j = 0; j < 4; ++j) acc[i][j] = (f32x8)0.0f;

  const int nk = Kd / BK;   // Kd is a multiple of 32 by construction
  fill_tiles(sA[0], sB[0], A, lda, B, bsk, bsn, m0, n0, 0, tid, brow);
  __syncthreads();

  for (int ki = 0; ki < nk; ++ki) {
    const int cur = ki & 1, nxt = cur ^ 1;

    if (ki + 2 < nk)   // speculative prefetch of the k+2 A tile
      __builtin_prefetch(A + (long)(m0 + (tid >> 1)) * lda + (ki + 2) * BK, 0, 1);

    if (ki + 1 < nk)   // stage next tile while WMMAs consume current one
      fill_tiles(sA[nxt], sB[nxt], A, lda, B, bsk, bsn, m0, n0, (ki + 1) * BK, tid, brow);

    // ---- fragment loads: 2x ds_load_b128 per frag, ISA 16-bit A/B layout ----
    bf16x16 afr[2], bfr[4];
#pragma unroll
    for (int tm = 0; tm < 2; ++tm) {
      const __bf16* p = &sA[cur][(wmi * 32 + tm * 16 + l16) * LSTR];
      bf16x8 lo = *(const bf16x8*)(p + hsel * 8);
      bf16x8 hi = *(const bf16x8*)(p + 16 + hsel * 8);
      afr[tm] = __builtin_shufflevector(lo, hi, 0,1,2,3,4,5,6,7,8,9,10,11,12,13,14,15);
    }
#pragma unroll
    for (int tn = 0; tn < 4; ++tn) {
      const __bf16* p = &sB[cur][(wni * 64 + tn * 16 + l16) * LSTR];
      bf16x8 lo = *(const bf16x8*)(p + hsel * 8);
      bf16x8 hi = *(const bf16x8*)(p + 16 + hsel * 8);
      bfr[tn] = __builtin_shufflevector(lo, hi, 0,1,2,3,4,5,6,7,8,9,10,11,12,13,14,15);
    }

#pragma unroll
    for (int tm = 0; tm < 2; ++tm)
#pragma unroll
      for (int tn = 0; tn < 4; ++tn)
        acc[tm][tn] = __builtin_amdgcn_wmma_f32_16x16x32_bf16(
            false, afr[tm], false, bfr[tn], (short)0, acc[tm][tn], false, false);

    __syncthreads();
  }

  // ---- epilogue: lane<16 -> n=lane rows 0..7 ; lane>=16 -> n=lane-16 rows 8..15
#pragma unroll
  for (int tm = 0; tm < 2; ++tm) {
#pragma unroll
    for (int tn = 0; tn < 4; ++tn) {
      int n  = n0 + wni * 64 + tn * 16 + l16;
      int mb = m0 + wmi * 32 + tm * 16 + hsel * 8;
      if (n < Nd) {
        float bv = bias ? bias[n] : 0.0f;
#pragma unroll
        for (int r = 0; r < 8; ++r) {
          int m = mb + r;
          float v = acc[tm][tn][r] * scale + bv;
          if (act == 1) v = gelu_t(v);
          if (resid) v += resid[(long)m * ldr + n];
          C[(long)m * ldc + n] = v;
        }
      }
    }
  }
}

// =====================================================================
// token coords (row/col/h/w per token) from grid_thw
// =====================================================================
__global__ void coords_k(const int* __restrict__ grid, int n_img,
                         int* __restrict__ rowi, int* __restrict__ coli,
                         int* __restrict__ hi, int* __restrict__ wi) {
  int t = blockIdx.x * blockDim.x + threadIdx.x;
  if (t >= NT) return;
  int cu = 0, local = t, h = 1, w = 1;
  for (int j = 0; j < n_img; ++j) {
    int hh = grid[j * 3 + 1], ww = grid[j * 3 + 2];
    int tp = grid[j * 3] * hh * ww;
    h = hh; w = ww; local = t - cu;
    if (t < cu + tp) break;
    cu += tp;
  }
  int hw = h * w;
  int spatial = local % hw;
  int mw = w / 2;
  int group = spatial / 4, intra = spatial % 4;
  rowi[t] = (group / mw) * 2 + intra / 2;
  coli[t] = (group % mw) * 2 + intra % 2;
  hi[t] = h; wi[t] = w;
}

// =====================================================================
// bilinear pos-embed interpolation -> pe[t][d]
// =====================================================================
__global__ void pe_k(const float* __restrict__ pos,
                     const int* __restrict__ rowi, const int* __restrict__ coli,
                     const int* __restrict__ hi, const int* __restrict__ wi,
                     float* __restrict__ pe) {
  int t = blockIdx.x;
  int h = hi[t], w = wi[t];
  float rf = (h > 1) ? (float)rowi[t] * (float)(GG - 1) / (float)((h - 1) > 1 ? (h - 1) : 1) : 0.0f;
  float cf = (w > 1) ? (float)coli[t] * (float)(GG - 1) / (float)((w - 1) > 1 ? (w - 1) : 1) : 0.0f;
  int r0 = (int)floorf(rf); int r1 = r0 + 1 < GG - 1 ? r0 + 1 : GG - 1;
  int c0 = (int)floorf(cf); int c1 = c0 + 1 < GG - 1 ? c0 + 1 : GG - 1;
  float wr = rf - (float)r0, wc = cf - (float)c0;
  const float* p00 = pos + (long)(r0 * GG + c0) * DM;
  const float* p01 = pos + (long)(r0 * GG + c1) * DM;
  const float* p10 = pos + (long)(r1 * GG + c0) * DM;
  const float* p11 = pos + (long)(r1 * GG + c1) * DM;
  float w00 = (1 - wr) * (1 - wc), w01 = (1 - wr) * wc, w10 = wr * (1 - wc), w11 = wr * wc;
  for (int i = threadIdx.x; i < DM; i += 256)
    pe[(long)t * DM + i] = p00[i] * w00 + p01[i] * w01 + p10[i] * w10 + p11[i] * w11;
}

// =====================================================================
// LayerNorm over D=1280
// =====================================================================
__global__ void layernorm_k(const float* __restrict__ x,
                            const float* __restrict__ s, const float* __restrict__ b,
                            float* __restrict__ out) {
  __shared__ float red[256];
  int t = blockIdx.x, tid = threadIdx.x;
  const float* row = x + (long)t * DM;
  float sum = 0.f, sq = 0.f;
  for (int i = tid; i < DM; i += 256) { float v = row[i]; sum += v; sq += v * v; }
  red[tid] = sum; __syncthreads();
  for (int o = 128; o > 0; o >>= 1) { if (tid < o) red[tid] += red[tid + o]; __syncthreads(); }
  float mu = red[0] / (float)DM; __syncthreads();
  red[tid] = sq; __syncthreads();
  for (int o = 128; o > 0; o >>= 1) { if (tid < o) red[tid] += red[tid + o]; __syncthreads(); }
  float var = red[0] / (float)DM - mu * mu;
  float inv = rsqrtf(var + 1e-6f);
  for (int i = tid; i < DM; i += 256)
    out[(long)t * DM + i] = (row[i] - mu) * inv * s[i] + b[i];
}

// =====================================================================
// QKV split + RoPE:  qkv[N][3][16][80] -> q/k/v [H][N][128] (zero-padded)
// =====================================================================
__global__ void rope_split_k(const float* __restrict__ qkv,
                             const int* __restrict__ rowi, const int* __restrict__ coli,
                             float* __restrict__ q, float* __restrict__ k,
                             float* __restrict__ v) {
  int t = blockIdx.x;
  int d = threadIdx.x;            // 0..127
  if (d >= KD) {                  // zero the padding columns 80..127
#pragma unroll 4
    for (int h = 0; h < NH; ++h) {
      long dst = ((long)h * NT + t) * KSTR + d;
      q[dst] = 0.0f; k[dst] = 0.0f; v[dst] = 0.0f;
    }
    return;
  }
  float rowf = (float)rowi[t], colf = (float)coli[t];
  int j = (d < 40) ? d : d - 40;
  float f;
  if (j < 20) f = rowf * powf(10000.0f, -(float)j / 20.0f);
  else        f = colf * powf(10000.0f, -(float)(j - 20) / 20.0f);
  float cs = cosf(f), sn = sinf(f);
  const float* base = qkv + (long)t * (3 * DM);
#pragma unroll 4
  for (int h = 0; h < NH; ++h) {
    long dst = ((long)h * NT + t) * KSTR + d;
    float qv = base[h * KD + d];
    float qr = (d < 40) ? -base[h * KD + d + 40] : base[h * KD + d - 40];
    q[dst] = qv * cs + qr * sn;
    float kv = base[DM + h * KD + d];
    float kr = (d < 40) ? -base[DM + h * KD + d + 40] : base[DM + h * KD + d - 40];
    k[dst] = kv * cs + kr * sn;
    v[dst] = base[2 * DM + h * KD + d];
  }
}

// =====================================================================
// row softmax over 512 (per-image key block -> mask implicit)
// =====================================================================
__global__ void softmax_k(float* __restrict__ S) {
  __shared__ float red[128];
  long r = blockIdx.x;
  float* row = S + r * IMG;
  int tid = threadIdx.x;
  float m = -3.4e38f;
  for (int i = tid; i < IMG; i += 128) m = fmaxf(m, row[i]);
  red[tid] = m; __syncthreads();
  for (int o = 64; o > 0; o >>= 1) { if (tid < o) red[tid] = fmaxf(red[tid], red[tid + o]); __syncthreads(); }
  m = red[0]; __syncthreads();
  float s = 0.f;
  for (int i = tid; i < IMG; i += 128) { float e = __expf(row[i] - m); row[i] = e; s += e; }
  red[tid] = s; __syncthreads();
  for (int o = 64; o > 0; o >>= 1) { if (tid < o) red[tid] += red[tid + o]; __syncthreads(); }
  float inv = 1.0f / red[0];
  for (int i = tid; i < IMG; i += 128) row[i] *= inv;
}

// =====================================================================
// host side
// =====================================================================
static void launch_gemm(hipStream_t st,
                        const float* A, long lda, const float* B, long bsk, long bsn,
                        const float* bias, const float* resid, long ldr,
                        float* C, long ldc, int Md, int Nd, int Kd,
                        float scale, int act,
                        int nbatch = 1, int nb2 = 1,
                        long ba1 = 0, long ba2 = 0, long bb1 = 0, long bb2 = 0,
                        long bc1 = 0, long bc2 = 0, long br1 = 0, long br2 = 0) {
  dim3 g((Nd + BN - 1) / BN, (Md + BM - 1) / BM, nbatch);
  hipLaunchKernelGGL(gemm_wmma_bf16, g, dim3(256), 0, st,
                     A, lda, B, bsk, bsn, bias, resid, ldr, C, ldc,
                     Nd, Kd, scale, act, nb2,
                     ba1, ba2, bb1, bb2, bc1, bc2, br1, br2);
}

extern "C" void kernel_launch(void* const* d_in, const int* in_sizes, int n_in,
                              void* d_out, int out_size, void* d_ws, size_t ws_size,
                              hipStream_t stream) {
  const float* pixels   = (const float*)d_in[0];
  const int*   grid_thw = (const int*)d_in[1];
  const float* pos_emb  = (const float*)d_in[2];
  const float* patch_w  = (const float*)d_in[3];
  const float* patch_b  = (const float*)d_in[4];
  const float* ln1_s    = (const float*)d_in[5];
  const float* ln1_b    = (const float*)d_in[6];
  const float* qkv_w    = (const float*)d_in[7];
  const float* qkv_b    = (const float*)d_in[8];
  const float* proj_w   = (const float*)d_in[9];
  const float* proj_b   = (const float*)d_in[10];
  const float* ln2_s    = (const float*)d_in[11];
  const float* ln2_b    = (const float*)d_in[12];
  const float* fc1_w    = (const float*)d_in[13];
  const float* fc1_b    = (const float*)d_in[14];
  const float* fc2_w    = (const float*)d_in[15];
  const float* fc2_b    = (const float*)d_in[16];
  const float* mn_s     = (const float*)d_in[17];
  const float* mn_b     = (const float*)d_in[18];
  const float* mfc1_w   = (const float*)d_in[19];
  const float* mfc1_b   = (const float*)d_in[20];
  const float* mfc2_w   = (const float*)d_in[21];
  const float* mfc2_b   = (const float*)d_in[22];
  float* out = (float*)d_out;
  int n_img = in_sizes[1] / 3;

  // ---- workspace carve (floats) ----
  float* x    = (float*)d_ws;                    // [N][D]
  float* hdn  = x    + (size_t)NT * DM;          // [N][D]
  float* qkvb = hdn  + (size_t)NT * DM;          // [N][3D]   (aliased as attn out o)
  float* q    = qkvb + (size_t)NT * 3 * DM;      // [H][N][128] zero-padded
  float* k    = q    + (size_t)NH * NT * KSTR;
  float* v    = k    + (size_t)NH * NT * KSTR;
  float* S    = v    + (size_t)NH * NT * KSTR;   // [H][N][512] (aliased: pe, ff)
  int* rowi = (int*)(S + (size_t)NH * NT * IMG);
  int* coli = rowi + NT;
  int* hi   = coli + NT;
  int* wi   = hi   + NT;
  float* o  = qkvb;
  float* pe = S;
  float* ff = S;

  const float inv_sqrt_k = 1.0f / sqrtf((float)KD);

  // ---- coords + pos embed ----
  hipLaunchKernelGGL(coords_k, dim3((NT + 255) / 256), dim3(256), 0, stream,
                     grid_thw, n_img, rowi, coli, hi, wi);
  hipLaunchKernelGGL(pe_k, dim3(NT), dim3(256), 0, stream,
                     pos_emb, rowi, coli, hi, wi, pe);

  // ---- patch embed: x = pixels @ patch_w + patch_b + pe ----
  launch_gemm(stream, pixels, 1536, patch_w, DM, 1, patch_b, pe, DM,
              x, DM, NT, DM, 1536, 1.0f, 0);

  // ---- transformer layers ----
  for (int l = 0; l < NL; ++l) {
    hipLaunchKernelGGL(layernorm_k, dim3(NT), dim3(256), 0, stream,
                       x, ln1_s + (size_t)l * DM, ln1_b + (size_t)l * DM, hdn);

    launch_gemm(stream, hdn, DM, qkv_w + (size_t)l * DM * 3 * DM, 3 * DM, 1,
                qkv_b + (size_t)l * 3 * DM, nullptr, 0,
                qkvb, 3 * DM, NT, 3 * DM, DM, 1.0f, 0);

    hipLaunchKernelGGL(rope_split_k, dim3(NT), dim3(128), 0, stream,
                       qkvb, rowi, coli, q, k, v);

    // scores[h][img]: (512x128pad) @ (128pad x 512) * 1/sqrt(80), batched 16*4
    launch_gemm(stream, q, KSTR, k, 1, KSTR, nullptr, nullptr, 0,
                S, IMG, IMG, IMG, KSTR, inv_sqrt_k, 0,
                NH * NIMG, NIMG,
                (long)NT * KSTR, (long)IMG * KSTR,    // A batch strides (h, img)
                (long)NT * KSTR, (long)IMG * KSTR,    // B batch strides
                (long)NT * IMG,  (long)IMG * IMG,     // C batch strides
                0, 0);

    hipLaunchKernelGGL(softmax_k, dim3(NH * NT), dim3(128), 0, stream, S);

    // o[img_tokens][h*80+d] = P (512x512) @ V (512x80, padded to 128), batched
    launch_gemm(stream, S, IMG, v, KSTR, 1, nullptr, nullptr, 0,
                o, DM, IMG, KD, IMG, 1.0f, 0,
                NH * NIMG, NIMG,
                (long)NT * IMG,  (long)IMG * IMG,     // A = S
                (long)NT * KSTR, (long)IMG * KSTR,    // B = v
                (long)KD, (long)IMG * DM,             // C = o
                0, 0);

    // x = x + o @ proj_w + proj_b
    launch_gemm(stream, o, DM, proj_w + (size_t)l * DM * DM, DM, 1,
                proj_b + (size_t)l * DM, x, DM,
                x, DM, NT, DM, DM, 1.0f, 0);

    hipLaunchKernelGGL(layernorm_k, dim3(NT), dim3(256), 0, stream,
                       x, ln2_s + (size_t)l * DM, ln2_b + (size_t)l * DM, hdn);

    // ff = gelu(hdn @ fc1 + b)
    launch_gemm(stream, hdn, DM, fc1_w + (size_t)l * DM * FFD, FFD, 1,
                fc1_b + (size_t)l * FFD, nullptr, 0,
                ff, FFD, NT, FFD, DM, 1.0f, 1);

    // x = x + ff @ fc2 + b
    launch_gemm(stream, ff, FFD, fc2_w + (size_t)l * FFD * DM, DM, 1,
                fc2_b + (size_t)l * DM, x, DM,
                x, DM, NT, DM, FFD, 1.0f, 0);
  }

  // ---- merger ----
  hipLaunchKernelGGL(layernorm_k, dim3(NT), dim3(256), 0, stream, x, mn_s, mn_b, hdn);
  // hdn viewed as (512, 5120)
  launch_gemm(stream, hdn, 4 * DM, mfc1_w, 4 * DM, 1, mfc1_b, nullptr, 0,
              ff, 4 * DM, NT / 4, 4 * DM, 4 * DM, 1.0f, 1);
  launch_gemm(stream, ff, 4 * DM, mfc2_w, OUTD, 1, mfc2_b, nullptr, 0,
              out, OUTD, NT / 4, OUTD, 4 * DM, 1.0f, 0);

  (void)ws_size; (void)out_size; (void)n_in;
}
